// LatticeLSTMLayer_sup_back_V0_40261023432820
// MI455X (gfx1250) — matrix operand order repaired and checked by reading, not compiled, not run on hardware
//
#include <hip/hip_runtime.h>
#include <stdint.h>

// ---------------- problem constants ----------------
#define B_    64
#define T_    512
#define X_    8
#define DC_   8
#define DW_   128
#define H_    128
#define H3_   384
#define KC_   256            // GEMM1 K = H (h_x) + DW (skip word)
#define NEGBIG 1e20f

#define NWG       8          // persistent workgroups, 8 batches each
#define NTHREADS  256        // 8 wave32 per WG
#define NWAVES    8

// ---------------- vector types ----------------
typedef _Float16 v8h  __attribute__((ext_vector_type(8)));
typedef _Float16 v16h __attribute__((ext_vector_type(16)));
typedef float    v8f  __attribute__((ext_vector_type(8)));
typedef unsigned int v4u __attribute__((ext_vector_type(4)));
typedef int      v8i  __attribute__((ext_vector_type(8)));
typedef int      v4i  __attribute__((ext_vector_type(4)));

#if defined(__has_builtin)
#  if __has_builtin(__builtin_amdgcn_tensor_load_to_lds) && __has_builtin(__builtin_amdgcn_s_wait_tensorcnt)
#    define HAVE_TDM 1
#  endif
#endif
#ifndef HAVE_TDM
#  define HAVE_TDM 0
#endif

// ---------------- LDS layout (bytes), 160 KB / WG ----------------
#define LOFF_A1   0        // f16 [64][256]  GEMM1 A tile (h_x || skip_word)     32 KB
#define LOFF_HST  32768    // f32 [64][128]  gathered h_x ; ALIASED as alpha f32 32 KB
#define LOFF_CST  65536    // f32 [64][128]  gathered c_x                        32 KB
#define LOFF_C1F  98304    // f32 [64][128]  c1_skip (f32, for merge)            32 KB
#define LOFF_C1H  131072   // f16 [64][128]  c1_skip (f16, GEMM3 A tile)         16 KB
#define LOFF_H0   147456   // f16 [16][128]  h0 tile (rows 8..15 zero)            4 KB
#define LOFF_GATE 151552   // f32 [3][8][128] i/o/g gates post-activation        12 KB
#define LDS_BYTES 163840

// ---------------- helpers ----------------
__device__ __forceinline__ float sigf(float x) { return 1.0f / (1.0f + __expf(-x)); }

// A-tile loader, 16x32 f16: lane<16 holds row M=lane, K = kc*32 + 0..15 contiguous;
// lane>=16 holds row M=lane-16, K = +16..31.  Two 16B loads -> v16h.
__device__ __forceinline__ v16h load_a16(const _Float16* base, int pitch, int r0, int kc, int lane) {
  const _Float16* p = base + (size_t)(r0 + (lane & 15)) * pitch + kc * 32 + ((lane >> 4) << 4);
  v8h lo = *(const v8h*)(p);
  v8h hi = *(const v8h*)(p + 8);
  return __builtin_shufflevector(lo, hi, 0,1,2,3,4,5,6,7,8,9,10,11,12,13,14,15);
}

// B-tile loader, 32x16 f16 (dense analogue of sparse B layout): lane<16 holds column N=lane,
// K = kc*32 + 0..15 contiguous; lane>=16 holds N=lane-16, K = +16..31.  One 32B load.
// Weights stored N-major ( W_t[n][k] = W[k][n] ), so this is a contiguous load.
__device__ __forceinline__ v16h load_b16(const _Float16* base, int pitch, int n0, int kc, int lane) {
  const _Float16* p = base + (size_t)(n0 + (lane & 15)) * pitch + kc * 32 + ((lane >> 4) << 4);
  return *(const v16h*)(p);
}

#define WMMA_F16(A, Bm, C) \
  __builtin_amdgcn_wmma_f32_16x16x32_f16(false, (A), false, (Bm), (short)0, (C), false, false)

// ---------------- kernel 1: one-time weight convert/transpose to f16 ----------------
// Wt_comb f16 [384][256]: k<128 -> w_hh_w[k][n], k>=128 -> w_ih_w[k-128][n]
// Whhc_t  f16 [384][128]: w_hh_c[k][n]
// Ahh_t   f16 [128][128]: a_hh[k][n]
__global__ void lattice_prep_weights(const float* __restrict__ w_hh_w,
                                     const float* __restrict__ w_ih_w,
                                     const float* __restrict__ w_hh_c,
                                     const float* __restrict__ a_hh,
                                     _Float16* __restrict__ Wt,
                                     _Float16* __restrict__ Whhc,
                                     _Float16* __restrict__ Ahh) {
  int i = blockIdx.x * blockDim.x + threadIdx.x;
  const int t1 = H3_ * KC_;            // 98304
  const int t2 = H3_ * H_;             // 49152
  const int t3 = H_ * H_;              // 16384
  if (i < t1) {
    int n = i / KC_, k = i % KC_;
    float v = (k < H_) ? w_hh_w[(size_t)k * H3_ + n] : w_ih_w[(size_t)(k - H_) * H3_ + n];
    Wt[(size_t)n * KC_ + k] = (_Float16)v;
  } else if (i < t1 + t2) {
    int j = i - t1; int n = j / H_, k = j % H_;
    Whhc[(size_t)n * H_ + k] = (_Float16)w_hh_c[(size_t)k * H3_ + n];
  } else if (i < t1 + t2 + t3) {
    int j = i - t1 - t2; int n = j / H_, k = j % H_;
    Ahh[(size_t)n * H_ + k] = (_Float16)a_hh[(size_t)k * H_ + n];
  }
}

// ---------------- kernel 2: persistent sequential lattice scan ----------------
__global__ void __launch_bounds__(NTHREADS, 1)
lattice_scan(float* __restrict__ out,                 // hs [B][T][H] then cs [B][T][H]
             const float* __restrict__ inp,           // [B][T][DC]
             const float* __restrict__ skip_words,    // [B][T][X][DW]
             const int*   __restrict__ skip_sources,  // [B][T][X]
             const int*   __restrict__ skip_count,    // [B][T]
             const float* __restrict__ w_ih_c,        // [DC][3H]
             const float* __restrict__ bias_c,        // [3H]
             const float* __restrict__ a_ih,          // [DC][H]
             const float* __restrict__ a_bias,        // [H]
             const float* __restrict__ bias_w,        // [3H]
             const _Float16* __restrict__ Wt,         // [384][256] N-major
             const _Float16* __restrict__ Whhc,       // [384][128] N-major
             const _Float16* __restrict__ Ahh,        // [128][128] N-major
             unsigned* __restrict__ bar) {
  extern __shared__ char smem[];
  _Float16* A1     = (_Float16*)(smem + LOFF_A1);
  float*    hstage = (float*)(smem + LOFF_HST);
  float*    alpha  = (float*)(smem + LOFF_HST);     // aliased: hstage dead after convert
  float*    cstage = (float*)(smem + LOFF_CST);
  float*    c1f32  = (float*)(smem + LOFF_C1F);
  _Float16* c1f16  = (_Float16*)(smem + LOFF_C1H);
  _Float16* h0t    = (_Float16*)(smem + LOFF_H0);
  float*    gates  = (float*)(smem + LOFF_GATE);

  const int wg   = blockIdx.x;
  const int tid  = threadIdx.x;
  const int wave = tid >> 5;
  const int lane = tid & 31;
  const int bWG  = wg * (B_ / NWG);                  // 8 batches per WG

  float* out_hs = out;
  float* out_cs = out + (size_t)B_ * T_ * H_;

  for (int t = 0; t < T_; ++t) {
    // ---------- phase A: gather h_x / c_x history rows (TDM gather-mode DMA) ----------
#if HAVE_TDM
    {
      // wave w gathers the 8 skip rows for batch bWG+w, for both h and c histories.
      const int b = bWG + wave;
      const int* sp = skip_sources + ((size_t)b * T_ + t) * X_;
      v4i idx;
      idx[0] = (sp[0] & 0xffff) | ((sp[1] & 0xffff) << 16);
      idx[1] = (sp[2] & 0xffff) | ((sp[3] & 0xffff) << 16);
      idx[2] = (sp[4] & 0xffff) | ((sp[5] & 0xffff) << 16);
      idx[3] = (sp[6] & 0xffff) | ((sp[7] & 0xffff) << 16);
      v4i g3 = {0, 0, 0, 0};
      v8i g4 = {0, 0, 0, 0, 0, 0, 0, 0};            // 6-arg builtin: extra group (zero-filled)

      v8i g1;                                       // D# group 1 (ISA 8.4)
      g1[0] = (2 << 16);                            // workgroup_mask=0, data_size=4B
      g1[1] = (H_ << 16);                           // tensor_dim0 = 128 (bits 79:48 low16)
      g1[2] = (T_ << 16);                           // tensor_dim1 = 512
      g1[3] = (H_ << 16);                           // tile_dim0 = 128 (row width)
      g1[4] = X_;                                   // tile_dim1 = #valid gather indices
      g1[5] = H_;                                   // tensor_dim0_stride = 128
      g1[6] = 0; g1[7] = 0;

      // h history gather -> hstage rows [wave*8 .. wave*8+7]
      {
        uint64_t ga = (uint64_t)(uintptr_t)(out_hs + (size_t)b * T_ * H_);
        v4u g0;
        g0[0] = 1u | (1u << 31);                    // count=1, gather_mode=1, 16-bit idx
        g0[1] = (unsigned)(uintptr_t)(hstage + wave * 8 * H_);   // lds_addr (bytes)
        g0[2] = (unsigned)ga;                       // global_addr[31:0]
        g0[3] = (unsigned)(ga >> 32) | (2u << 30);  // global_addr[56:32] | type=2
        __builtin_amdgcn_tensor_load_to_lds(g0, g1, idx, g3, g4, 0);
      }
      // c history gather -> cstage rows [wave*8 .. wave*8+7]
      {
        uint64_t ga = (uint64_t)(uintptr_t)(out_cs + (size_t)b * T_ * H_);
        v4u g0;
        g0[0] = 1u | (1u << 31);
        g0[1] = (unsigned)(uintptr_t)(cstage + wave * 8 * H_);
        g0[2] = (unsigned)ga;
        g0[3] = (unsigned)(ga >> 32) | (2u << 30);
        __builtin_amdgcn_tensor_load_to_lds(g0, g1, idx, g3, g4, 0);
      }
      __builtin_amdgcn_s_wait_tensorcnt(0);
    }
#else
    for (int e = tid; e < 64 * H_; e += NTHREADS) {
      int r = e >> 7, h = e & (H_ - 1);
      int b = bWG + (r >> 3), x = r & 7;
      int src = skip_sources[((size_t)b * T_ + t) * X_ + x];     // hist slot src+1 == hs[.., src]
      hstage[r * H_ + h] = out_hs[((size_t)b * T_ + src) * H_ + h];
      cstage[r * H_ + h] = out_cs[((size_t)b * T_ + src) * H_ + h];
    }
#endif
    __syncthreads();

    // ---------- phase B: build f16 A-tiles (h_x || skip_word), h0 tile ----------
    for (int e = tid; e < 64 * KC_; e += NTHREADS) {
      int r = e >> 8, k = e & (KC_ - 1);
      float v;
      if (k < H_) {
        v = hstage[r * H_ + k];
      } else {
        int b = bWG + (r >> 3), x = r & 7;
        const float* swp = skip_words + (((size_t)b * T_ + t) * X_ + x) * DW_ + (k - H_);
        v = *swp;
        if (t + 1 < T_ && (((k - H_) & 15) == 0))   // prefetch next step's skip word rows
          __builtin_prefetch(swp + (size_t)X_ * DW_, 0, 1);
      }
      A1[r * KC_ + k] = (_Float16)v;
    }
    for (int e = tid; e < 16 * H_; e += NTHREADS) {
      int m = e >> 7, h = e & (H_ - 1);
      float v = 0.0f;
      if (m < 8 && t > 0) v = out_hs[((size_t)(bWG + m) * T_ + (t - 1)) * H_ + h];
      h0t[m * H_ + h] = (_Float16)v;
    }
    __syncthreads();

    // ---------- phase C: GEMM1 (gw, fused WordLSTM activation) + GEMM2 (gc gates) ----------
    // GEMM1: M=64 rows (this WG), N=384 as 8 h-tiles x {f,i,g} triple, K=256.
    for (int u = wave * 4; u < wave * 4 + 4; ++u) {
      const int rt = u >> 3, ht = u & 7;
      const int r0 = rt * 16, h0c = ht * 16;
      v8f accF = {0,0,0,0,0,0,0,0}, accI = {0,0,0,0,0,0,0,0}, accG = {0,0,0,0,0,0,0,0};
#pragma unroll
      for (int kc = 0; kc < KC_ / 32; ++kc) {
        v16h a  = load_a16(A1, KC_, r0, kc, lane);
        v16h bf = load_b16(Wt, KC_, 0 * H_ + h0c, kc, lane);
        v16h bi = load_b16(Wt, KC_, 1 * H_ + h0c, kc, lane);
        v16h bg = load_b16(Wt, KC_, 2 * H_ + h0c, kc, lane);
        accF = WMMA_F16(a, bf, accF);
        accI = WMMA_F16(a, bi, accI);
        accG = WMMA_F16(a, bg, accG);
      }
      const int hi = lane >> 4, h = h0c + (lane & 15);
      const float bf0 = bias_w[h], bf1 = bias_w[H_ + h], bf2 = bias_w[2 * H_ + h];
#pragma unroll
      for (int j = 0; j < 8; ++j) {
        int r = r0 + j + hi * 8;
        float cx = cstage[r * H_ + h];
        float c1 = sigf(accF[j] + bf0) * cx + sigf(accI[j] + bf1) * tanhf(accG[j] + bf2);
        c1f32[r * H_ + h] = c1;
        c1f16[r * H_ + h] = (_Float16)c1;
      }
    }
    // GEMM2: M=16 (8 real batch rows), N=384 as 8 h-tiles x {i,o,g}, K=128. One h-tile per wave.
    {
      const int h0c = wave * 16;
      v8f aI = {0,0,0,0,0,0,0,0}, aO = {0,0,0,0,0,0,0,0}, aG = {0,0,0,0,0,0,0,0};
#pragma unroll
      for (int kc = 0; kc < H_ / 32; ++kc) {
        v16h a  = load_a16(h0t, H_, 0, kc, lane);
        v16h bI = load_b16(Whhc, H_, 0 * H_ + h0c, kc, lane);
        v16h bO = load_b16(Whhc, H_, 1 * H_ + h0c, kc, lane);
        v16h bG = load_b16(Whhc, H_, 2 * H_ + h0c, kc, lane);
        aI = WMMA_F16(a, bI, aI);
        aO = WMMA_F16(a, bO, aO);
        aG = WMMA_F16(a, bG, aG);
      }
      const int hi = lane >> 4, h = h0c + (lane & 15);
#pragma unroll
      for (int j = 0; j < 8; ++j) {
        int m = j + hi * 8;
        if (m < 8) {
          int b = bWG + m;
          const float* ip = inp + ((size_t)b * T_ + t) * DC_;
          float w0 = bias_c[h], w1 = bias_c[H_ + h], w2 = bias_c[2 * H_ + h];
#pragma unroll
          for (int d = 0; d < DC_; ++d) {
            float v = ip[d];
            w0 += v * w_ih_c[d * H3_ + h];
            w1 += v * w_ih_c[d * H3_ + H_ + h];
            w2 += v * w_ih_c[d * H3_ + 2 * H_ + h];
          }
          gates[(0 * 8 + m) * H_ + h] = sigf(aI[j] + w0);    // i_g (sigmoid)
          gates[(1 * 8 + m) * H_ + h] = sigf(aO[j] + w1);    // o_g (sigmoid)
          gates[(2 * 8 + m) * H_ + h] = tanhf(aG[j] + w2);   // g_g (tanh)
        }
      }
    }
    __syncthreads();

    // ---------- phase D: GEMM3 (alpha = sigmoid(a_wi + c1_skip @ a_hh)) ----------
    for (int u = wave * 4; u < wave * 4 + 4; ++u) {
      const int rt = u >> 3, nt = u & 7;
      const int r0 = rt * 16, n0c = nt * 16;
      v8f acc = {0,0,0,0,0,0,0,0};
#pragma unroll
      for (int kc = 0; kc < H_ / 32; ++kc) {
        v16h a  = load_a16(c1f16, H_, r0, kc, lane);
        v16h bb = load_b16(Ahh, H_, n0c, kc, lane);
        acc = WMMA_F16(a, bb, acc);
      }
      const int hi = lane >> 4, h = n0c + (lane & 15);
#pragma unroll
      for (int j = 0; j < 8; ++j) {
        int r = r0 + j + hi * 8;
        int b = bWG + (r >> 3);
        const float* ip = inp + ((size_t)b * T_ + t) * DC_;
        float aw = a_bias[h];
#pragma unroll
        for (int d = 0; d < DC_; ++d) aw += ip[d] * a_ih[d * H_ + h];
        alpha[r * H_ + h] = sigf(acc[j] + aw);
      }
    }
    __syncthreads();

    // ---------- phase E: masked softmax merge over {g} U skip cells, write outputs ----------
    for (int e = tid; e < 8 * H_; e += NTHREADS) {
      int m = e >> 7, h = e & (H_ - 1);
      int b = bWG + m;
      int cnt = skip_count[(size_t)b * T_ + t];
      float gi = gates[(0 * 8 + m) * H_ + h];
      float go = gates[(1 * 8 + m) * H_ + h];
      float gg = gates[(2 * 8 + m) * H_ + h];
      float s = __expf(gi);
      float acc = s * gg;
#pragma unroll
      for (int x = 0; x < X_; ++x) {
        int r = m * X_ + x;
        float a = alpha[r * H_ + h];
        if (x >= cnt) a -= NEGBIG;                   // mask trick -> exp() == 0
        float ee = __expf(a);
        s += ee;
        acc += ee * c1f32[r * H_ + h];
      }
      float c1 = acc / s;
      float h1 = go * tanhf(c1);
      out_hs[((size_t)b * T_ + t) * H_ + h] = h1;
      out_cs[((size_t)b * T_ + t) * H_ + h] = c1;
    }

    // ---------- phase F: device-wide step barrier ----------
    __threadfence();
    __syncthreads();
#if __has_builtin(__builtin_amdgcn_s_cluster_barrier)
    __builtin_amdgcn_s_cluster_barrier();            // NOP outside cluster dispatch; atomic
#endif                                               // barrier below is the authoritative sync
    if (tid == 0) {
      __hip_atomic_fetch_add(bar, 1u, __ATOMIC_RELEASE, __HIP_MEMORY_SCOPE_AGENT);
      const unsigned target = (unsigned)NWG * (unsigned)(t + 1);
      while (__hip_atomic_load(bar, __ATOMIC_ACQUIRE, __HIP_MEMORY_SCOPE_AGENT) < target)
        __builtin_amdgcn_s_sleep(2);
    }
    __syncthreads();
    __threadfence();
  }
}

// ---------------- host-side launch ----------------
extern "C" void kernel_launch(void* const* d_in, const int* in_sizes, int n_in,
                              void* d_out, int out_size, void* d_ws, size_t ws_size,
                              hipStream_t stream) {
  (void)in_sizes; (void)n_in; (void)out_size; (void)ws_size;
  const float* inp          = (const float*)d_in[0];
  const float* skip_words   = (const float*)d_in[1];
  /* d_in[2] seq_len: unused (T static) */
  const int*   skip_sources = (const int*)d_in[3];
  const int*   skip_count   = (const int*)d_in[4];
  const float* w_ih_c       = (const float*)d_in[5];
  const float* w_hh_c       = (const float*)d_in[6];
  const float* bias_c       = (const float*)d_in[7];
  const float* a_ih         = (const float*)d_in[8];
  const float* a_hh         = (const float*)d_in[9];
  const float* a_bias       = (const float*)d_in[10];
  const float* w_ih_w       = (const float*)d_in[11];
  const float* w_hh_w       = (const float*)d_in[12];
  const float* bias_w       = (const float*)d_in[13];

  char* ws = (char*)d_ws;
  _Float16* Wt   = (_Float16*)(ws);                        // 384*256*2 = 196608 B
  _Float16* Whhc = (_Float16*)(ws + 196608);               // 384*128*2 =  98304 B
  _Float16* Ahh  = (_Float16*)(ws + 196608 + 98304);       // 128*128*2 =  32768 B
  unsigned* bar  = (unsigned*)(ws + 196608 + 98304 + 32768);

  (void)hipMemsetAsync(bar, 0, 128, stream);

  lattice_prep_weights<<<640, 256, 0, stream>>>(w_hh_w, w_ih_w, w_hh_c, a_hh, Wt, Whhc, Ahh);

  lattice_scan<<<NWG, NTHREADS, LDS_BYTES, stream>>>(
      (float*)d_out, inp, skip_words, skip_sources, skip_count,
      w_ih_c, bias_c, a_ih, a_bias, bias_w, Wt, Whhc, Ahh, bar);
}
